// QuantizedLinear_43430709297290
// MI455X (gfx1250) — compile-verified, matrix operand order checked by reading, not run on hardware
//
#include <hip/hip_runtime.h>

typedef __attribute__((ext_vector_type(16))) _Float16 v16h;
typedef __attribute__((ext_vector_type(8)))  _Float16 v8h;
typedef __attribute__((ext_vector_type(4)))  _Float16 v4h;
typedef __attribute__((ext_vector_type(8)))  float    v8f;
typedef __attribute__((ext_vector_type(4)))  float    v4f;
typedef __attribute__((ext_vector_type(4)))  int      v4i;

#define TM 128
#define TN 128
#define TK 32
#define LDA 40   // halves per LDS row (32 data + 8 pad -> 80B stride, bank-conflict free b128)
#define LDB 40

#define COEF_SCALE 0.02f
#define COEF_ZERO  128.0f
#define INT_SCALE  0.01f
#define INT_ZERO   128.0f

__global__ __launch_bounds__(256)
void QuantizedLinear_43430709297290_kernel(const float* __restrict__ X,
                                           const int*   __restrict__ Q,
                                           const int*   __restrict__ Bq,
                                           float*       __restrict__ Out,
                                           int M, int N, int K) {
  __shared__ __align__(16) _Float16 As[2][TM * LDA]; // x tile, f16, row-major [m][k]
  __shared__ __align__(16) _Float16 Bs[2][TN * LDB]; // (q-128) tile, EXACT f16, transposed [n][k]

  const int t    = threadIdx.x;
  const int lane = t & 31;
  const int wid  = t >> 5;        // 0..7
  const int mw   = wid >> 1;      // 0..3 : wave row block (32 rows)
  const int nw   = wid & 1;       // 0..1 : wave col block (64 cols)
  const int l15  = lane & 15;
  const int hi   = lane >> 4;     // 0: K 0-7/16-23 halves, 1: K 8-15/24-31
  const int m_base = blockIdx.y * TM;
  const int n_base = blockIdx.x * TN;

  const size_t Ks = (size_t)K, Ns = (size_t)N;
  const int KT = K / TK;

  const v8f vzero = {};
  v8f acc[2][4];
#pragma unroll
  for (int i = 0; i < 2; ++i)
#pragma unroll
    for (int j = 0; j < 4; ++j)
      acc[i][j] = vzero;

  // Cooperative-load index precompute:
  //   x chunks: c = i*256+t -> row = c>>3 (0..127), kc = c&7 (4 floats each)
  //   q chunks: c = i*256+t -> kr  = c>>5 (0..31),  nc = c&31 (4 ints each)
  int xrow[4], xkc[4], qkr[4], qnc[4];
#pragma unroll
  for (int i = 0; i < 4; ++i) {
    const int c = i * 256 + t;
    xrow[i] = c >> 3; xkc[i] = c & 7;
    qkr[i]  = c >> 5; qnc[i] = c & 31;
  }

  v4f xreg[4];
  v4i qreg[4];

  // ---- prologue: global-load tile 0, convert, stage into LDS buf 0 ----
#pragma unroll
  for (int i = 0; i < 4; ++i) {
    xreg[i] = *(const v4f*)(X + (size_t)(m_base + xrow[i]) * Ks + (size_t)(xkc[i] * 4));
    qreg[i] = *(const v4i*)(Q + (size_t)qkr[i] * Ns + (size_t)(n_base + qnc[i] * 4));
  }
#pragma unroll
  for (int i = 0; i < 4; ++i) {
    v4h h;
#pragma unroll
    for (int u = 0; u < 4; ++u) h[u] = (_Float16)xreg[i][u];
    *(v4h*)&As[0][xrow[i] * LDA + xkc[i] * 4] = h;
    // (q-128) is an integer in [-128,127]: EXACT in f16. Scale folded into epilogue.
#pragma unroll
    for (int u = 0; u < 4; ++u)
      Bs[0][(qnc[i] * 4 + u) * LDB + qkr[i]] =
          (_Float16)((float)qreg[i][u] - COEF_ZERO);
  }

  int buf = 0;
  for (int kt = 0; kt < KT; ++kt) {
    __syncthreads(); // LDS[buf] ready for all waves; previous compute of other buf done

    const bool more = (kt + 1) < KT;
    if (more) {
      const int k0 = (kt + 1) * TK;
#pragma unroll
      for (int i = 0; i < 4; ++i) {
        xreg[i] = *(const v4f*)(X + (size_t)(m_base + xrow[i]) * Ks + (size_t)(k0 + xkc[i] * 4));
        qreg[i] = *(const v4i*)(Q + (size_t)(k0 + qkr[i]) * Ns + (size_t)(n_base + qnc[i] * 4));
      }
    }

    // ---- load A/B fragments from LDS in the WMMA wave32 lane layout ----
    // lane<16 : halves[0..7]=K0..7,  halves[8..15]=K16..23  (chunk offsets +0, +16 halves)
    // lane>=16: halves[0..7]=K8..15, halves[8..15]=K24..31  (chunk offsets +8, +24 halves)
    v16h a[2], b[4];
#pragma unroll
    for (int i = 0; i < 2; ++i) {
      const _Float16* p = &As[buf][(mw * 32 + i * 16 + l15) * LDA + hi * 8];
      const v8h lo = *(const v8h*)p;
      const v8h hh = *(const v8h*)(p + 16);
      a[i] = __builtin_shufflevector(lo, hh, 0,1,2,3,4,5,6,7,8,9,10,11,12,13,14,15);
    }
#pragma unroll
    for (int j = 0; j < 4; ++j) {
      const _Float16* p = &Bs[buf][(nw * 64 + j * 16 + l15) * LDB + hi * 8];
      const v8h lo = *(const v8h*)p;
      const v8h hh = *(const v8h*)(p + 16);
      b[j] = __builtin_shufflevector(lo, hh, 0,1,2,3,4,5,6,7,8,9,10,11,12,13,14,15);
    }

#pragma unroll
    for (int i = 0; i < 2; ++i)
#pragma unroll
      for (int j = 0; j < 4; ++j)
        acc[i][j] = __builtin_amdgcn_wmma_f32_16x16x32_f16(
            /*neg_a=*/false, a[i], /*neg_b=*/false, b[j],
            /*c_mod=*/(short)0, acc[i][j],
            /*reuse_a=*/false, /*reuse_b=*/false);

    if (more) {
      const int nb = buf ^ 1;
#pragma unroll
      for (int i = 0; i < 4; ++i) {
        v4h h;
#pragma unroll
        for (int u = 0; u < 4; ++u) h[u] = (_Float16)xreg[i][u];
        *(v4h*)&As[nb][xrow[i] * LDA + xkc[i] * 4] = h;
#pragma unroll
        for (int u = 0; u < 4; ++u)
          Bs[nb][(qnc[i] * 4 + u) * LDB + qkr[i]] =
              (_Float16)((float)qreg[i][u] - COEF_ZERO);
      }
    }
    buf ^= 1;
  }

  // ---- epilogue: out = COEF_SCALE*acc + bias, store f32 per C/D VGPR layout ----
  // VGPR r of a 16x16 f32 tile: row = r + 8*(lane>=16), col = lane&15
#pragma unroll
  for (int j = 0; j < 4; ++j) {
    const int n = n_base + nw * 64 + j * 16 + l15;
    const float bias = INT_SCALE * ((float)Bq[n] - INT_ZERO);
#pragma unroll
    for (int i = 0; i < 2; ++i) {
      const int mrow = m_base + mw * 32 + i * 16 + hi * 8;
#pragma unroll
      for (int r = 0; r < 8; ++r) {
        Out[(size_t)(mrow + r) * Ns + n] = fmaf(COEF_SCALE, acc[i][j][r], bias);
      }
    }
  }
}

extern "C" void kernel_launch(void* const* d_in, const int* in_sizes, int n_in,
                              void* d_out, int out_size, void* d_ws, size_t ws_size,
                              hipStream_t stream) {
  const float* X  = (const float*)d_in[0];   // [M, K] fp32
  const int*   Q  = (const int*)d_in[1];     // [K, N] uint8-valued int32
  const int*   Bq = (const int*)d_in[2];     // [N]    uint8-valued int32
  float* Out = (float*)d_out;                // [M, N] fp32

  const int N = in_sizes[2];
  const int K = in_sizes[1] / N;
  const int M = in_sizes[0] / K;

  dim3 grid(N / TN, M / TM);   // (32, 64) for the reference shapes
  QuantizedLinear_43430709297290_kernel<<<grid, dim3(256), 0, stream>>>(
      X, Q, Bq, Out, M, N, K);
}
